// LSTMModel_42795054137616
// MI455X (gfx1250) — compile-verified, matrix operand order checked by reading, not run on hardware
//
#include <hip/hip_runtime.h>
#include <hip/hip_bf16.h>
#include <stddef.h>

// ---------------------------------------------------------------------------
// LSTM (B=64, T=512, D=512, U=1024), return last hidden state [64,1024] f32.
//
// Phase 1: xW = bf16(x) @ bf16(W) + b  -> bf16 [B*T, 4096] via v_wmma bf16.
// Phase 2: 512 sequential step kernels: z = xW[t] + h @ R (bf16 WMMA, f32 acc),
//          gates, c/h update. h double-buffered bf16; c kept f32.
// ---------------------------------------------------------------------------

typedef __attribute__((ext_vector_type(16))) __bf16 v16bf;
typedef __attribute__((ext_vector_type(8)))  float  v8f;

#define BATCH 64
#define TSTEPS 512
#define DIN 512
#define UNITS 1024
#define GDIM 4096            // 4*UNITS
#define ROWS (BATCH * TSTEPS) // 32768

// -------- workspace layout (bytes, 256-aligned sections) --------
#define OFF_XB  ((size_t)0)                       // x bf16   [32768,512]   33.5MB
#define OFF_WT  ((size_t)33554432)                // Wt bf16  [4096,512]     4.2MB
#define OFF_RT  ((size_t)37748736)                // Rt bf16  [4096,1024]    8.4MB
#define OFF_XW  ((size_t)46137344)                // xW bf16  [32768,4096] 268.4MB
#define OFF_H0  ((size_t)314572800)               // h0 bf16  [64,1024]
#define OFF_H1  ((size_t)314703872)               // h1 bf16  [64,1024]
#define OFF_C   ((size_t)314834944)               // c  f32   [64,1024]

__device__ __forceinline__ unsigned short f32_to_bf16_bits(float f) {
    unsigned int u = __float_as_uint(f);
    u = (u + 0x7FFFu + ((u >> 16) & 1u)) >> 16;   // round-to-nearest-even
    return (unsigned short)u;
}
__device__ __forceinline__ float bf16_bits_to_f32(unsigned short s) {
    return __uint_as_float(((unsigned int)s) << 16);
}
__device__ __forceinline__ float fast_sigmoid(float x) {
    return 1.0f / (1.0f + __expf(-x));
}
__device__ __forceinline__ float fast_tanh(float x) {
    float e = __expf(2.0f * x);
    return (e - 1.0f) / (e + 1.0f);
}

// -------- elementwise f32 -> bf16 copy (same layout) --------
__global__ void convert_bf16_kernel(const float* __restrict__ src,
                                    unsigned short* __restrict__ dst, int n) {
    for (int i = blockIdx.x * blockDim.x + threadIdx.x; i < n;
         i += gridDim.x * blockDim.x)
        dst[i] = f32_to_bf16_bits(src[i]);
}

// -------- transpose + convert: src[rows][cols] f32 -> dst[cols][rows] bf16 --
__global__ void transpose_bf16_kernel(const float* __restrict__ src,
                                      unsigned short* __restrict__ dst,
                                      int rows, int cols) {
    int total = rows * cols;
    for (int i = blockIdx.x * blockDim.x + threadIdx.x; i < total;
         i += gridDim.x * blockDim.x) {
        int r = i / cols, c = i - r * cols;
        dst[(size_t)c * rows + r] = f32_to_bf16_bits(src[i]);
    }
}

// -------- zero h0/h1 (bf16) and c (f32) --------
__global__ void init_state_kernel(unsigned short* __restrict__ h0,
                                  unsigned short* __restrict__ h1,
                                  float* __restrict__ c) {
    int i = blockIdx.x * blockDim.x + threadIdx.x;   // 65536 threads
    h0[i] = 0; h1[i] = 0; c[i] = 0.0f;
}

// ---------------------------------------------------------------------------
// Phase 1 GEMM: xW[m][n] = sum_k xb[m][k]*Wt[n][k] + bias[n], bf16 output.
// One wave per 16x16 output tile, K=512 -> 16 x v_wmma_f32_16x16x32_bf16.
// ---------------------------------------------------------------------------
__global__ void gemm_xw_kernel(const unsigned short* __restrict__ xb,   // [32768,512]
                               const unsigned short* __restrict__ Wt,   // [4096,512]
                               const float* __restrict__ bias,          // [4096]
                               unsigned short* __restrict__ xW) {       // [32768,4096]
    const int lane  = threadIdx.x & 31;
    const int wave  = threadIdx.x >> 5;
    const int gw    = blockIdx.x * (blockDim.x >> 5) + wave;  // 524288 tiles exact
    const int NT    = GDIM / 16;                               // 256
    const int mtile = gw / NT;
    const int ntile = gw - mtile * NT;
    const int m0 = mtile * 16, n0 = ntile * 16;

    const int lrow  = lane & 15;          // A row / B column within tile
    const int khalf = (lane >> 4) * 16;   // K half: contiguous k = 16h + s, same for A and B

    const unsigned short* arow = xb + (size_t)(m0 + lrow) * DIN + khalf;
    const unsigned short* brow = Wt + (size_t)(n0 + lrow) * DIN + khalf;

    v8f acc = {};
    #pragma unroll
    for (int k = 0; k < DIN; k += 32) {
        v16bf a = *(const v16bf*)(arow + k);
        v16bf b = *(const v16bf*)(brow + k);
        acc = __builtin_amdgcn_wmma_f32_16x16x32_bf16(
            false, a, false, b, (short)0, acc, false, false);
    }

    const int n  = n0 + (lane & 15);
    const float bv = bias[n];
    const int mh = (lane >> 4) << 3;      // lanes 16-31 hold M = r+8
    #pragma unroll
    for (int r = 0; r < 8; ++r) {
        int m = m0 + r + mh;
        xW[(size_t)m * GDIM + n] = f32_to_bf16_bits(acc[r] + bv);
    }
}

// ---------------------------------------------------------------------------
// Phase 2 step: one block = one 16x16 (batch x unit) tile. Wave g (0..3)
// computes gate g's z over full K=1024 (32 WMMAs), z exchanged via LDS,
// wave 0 applies gate nonlinearities and updates c / h.
// grid = 256 blocks x 128 threads.
// ---------------------------------------------------------------------------
__global__ void lstm_step_kernel(const unsigned short* __restrict__ xW,   // [32768,4096]
                                 const unsigned short* __restrict__ Rt,   // [4096,1024]
                                 const unsigned short* __restrict__ hin,  // [64,1024] bf16
                                 unsigned short* __restrict__ hout,       // [64,1024] bf16
                                 float* __restrict__ c,                   // [64,1024]
                                 float* __restrict__ out,                 // [64,1024]
                                 int t, int isLast) {
    __shared__ float zsh[4][16][16];

    const int lane = threadIdx.x & 31;
    const int gate = threadIdx.x >> 5;            // wave id == gate id
    const int NT   = UNITS / 16;                  // 64
    const int mtile = blockIdx.x / NT;
    const int ntile = blockIdx.x - mtile * NT;
    const int m0 = mtile * 16, n0 = ntile * 16;

    const int lrow  = lane & 15;
    const int khalf = (lane >> 4) * 16;

    const unsigned short* arow = hin + (size_t)(m0 + lrow) * UNITS + khalf;
    const unsigned short* brow =
        Rt + (size_t)(gate * UNITS + n0 + lrow) * UNITS + khalf;

    v8f acc = {};
    #pragma unroll 4
    for (int k = 0; k < UNITS; k += 32) {
        v16bf a = *(const v16bf*)(arow + k);
        v16bf b = *(const v16bf*)(brow + k);
        acc = __builtin_amdgcn_wmma_f32_16x16x32_bf16(
            false, a, false, b, (short)0, acc, false, false);
    }

    // z = h@R (acc) + xW[b*T + t][gate*U + n]; scatter to LDS in (m,n) order
    const int ncol = gate * UNITS + n0 + (lane & 15);
    const int mh   = (lane >> 4) << 3;
    #pragma unroll
    for (int r = 0; r < 8; ++r) {
        int m = r + mh;
        float xw = bf16_bits_to_f32(
            xW[(size_t)((m0 + m) * TSTEPS + t) * GDIM + ncol]);
        zsh[gate][m][lane & 15] = acc[r] + xw;
    }
    __syncthreads();

    // elementwise epilogue on wave 0: 256 elements, 8 per lane
    if (gate == 0) {
        #pragma unroll
        for (int e = lane; e < 256; e += 32) {
            int m = e >> 4, n = e & 15;
            float zi = zsh[0][m][n];
            float zf = zsh[1][m][n];
            float zg = zsh[2][m][n];
            float zo = zsh[3][m][n];
            float ig = fast_sigmoid(zi);
            float fg = fast_sigmoid(zf);
            float gg = fast_tanh(zg);
            float og = fast_sigmoid(zo);
            size_t idx = (size_t)(m0 + m) * UNITS + (n0 + n);
            float cn = fg * c[idx] + ig * gg;
            c[idx] = cn;
            float hn = og * fast_tanh(cn);
            hout[idx] = f32_to_bf16_bits(hn);
            if (isLast) out[idx] = hn;
        }
    }
}

// ---------------------------------------------------------------------------
extern "C" void kernel_launch(void* const* d_in, const int* in_sizes, int n_in,
                              void* d_out, int out_size, void* d_ws, size_t ws_size,
                              hipStream_t stream) {
    const float* x    = (const float*)d_in[0];   // [64,512,512]
    const float* W    = (const float*)d_in[1];   // [512,4096]
    const float* R    = (const float*)d_in[2];   // [1024,4096]
    const float* bias = (const float*)d_in[3];   // [4096]
    float* out = (float*)d_out;                  // [64,1024]

    char* ws = (char*)d_ws;
    unsigned short* xb  = (unsigned short*)(ws + OFF_XB);
    unsigned short* Wt  = (unsigned short*)(ws + OFF_WT);
    unsigned short* Rt  = (unsigned short*)(ws + OFF_RT);
    unsigned short* xW  = (unsigned short*)(ws + OFF_XW);
    unsigned short* h0  = (unsigned short*)(ws + OFF_H0);
    unsigned short* h1  = (unsigned short*)(ws + OFF_H1);
    float*          cst = (float*)(ws + OFF_C);

    // ---- conversions / transposes (one-time per call) ----
    convert_bf16_kernel<<<4096, 256, 0, stream>>>(x, xb, ROWS * DIN);
    transpose_bf16_kernel<<<2048, 256, 0, stream>>>(W, Wt, DIN, GDIM);
    transpose_bf16_kernel<<<4096, 256, 0, stream>>>(R, Rt, UNITS, GDIM);
    init_state_kernel<<<BATCH * UNITS / 256, 256, 0, stream>>>(h0, h1, cst);

    // ---- phase 1: big WMMA GEMM, 524288 wave-tiles / 8 waves per block ----
    gemm_xw_kernel<<<65536, 256, 0, stream>>>(xb, Wt, bias, xW);

    // ---- phase 2: 512 sequential recurrence steps (graph-captured) ----
    for (int t = 0; t < TSTEPS; ++t) {
        unsigned short* hin  = (t & 1) ? h1 : h0;
        unsigned short* hout = (t & 1) ? h0 : h1;
        lstm_step_kernel<<<(BATCH / 16) * (UNITS / 16), 128, 0, stream>>>(
            xW, Rt, hin, hout, cst, out, t, (t == TSTEPS - 1) ? 1 : 0);
    }
}
